// HeteroCompGCN_52183852646756
// MI455X (gfx1250) — compile-verified, matrix operand order checked by reading, not run on hardware
//
#include <hip/hip_runtime.h>

typedef __attribute__((ext_vector_type(16))) __bf16 v16bf;
typedef __attribute__((ext_vector_type(8)))  float  v8f;
typedef __attribute__((ext_vector_type(4)))  int    i4v;

#define FC 128   // feature width for every layer (C1 = H = OUT = 128)

#if defined(__HIP_DEVICE_COMPILE__) && defined(__has_builtin)
#if __has_builtin(__builtin_amdgcn_global_load_async_to_lds_b128)
#define HAVE_ASYNC_LDS 1
#endif
#endif

#ifdef HAVE_ASYNC_LDS
typedef __attribute__((address_space(1))) i4v* gas_i4p;
typedef __attribute__((address_space(3))) i4v* las_i4p;
#endif

__device__ __forceinline__ void wait_async_zero() {
#if defined(__HIP_DEVICE_COMPILE__)
#if defined(__has_builtin) && __has_builtin(__builtin_amdgcn_s_wait_asynccnt)
  __builtin_amdgcn_s_wait_asynccnt(0);
#else
  asm volatile("s_wait_asynccnt 0x0" ::: "memory");
#endif
#endif
}

// ---------------------------------------------------------------------------
// h0[i, :] = concat(x[i, 0:96], type_emb[node_type_ids[i], 0:32])
// ---------------------------------------------------------------------------
__global__ void k_build_h0(const float* __restrict__ x,
                           const int* __restrict__ tid,
                           const float* __restrict__ temb,
                           float* __restrict__ h0, int N) {
  long i = (long)blockIdx.x * blockDim.x + threadIdx.x;
  long total = (long)N * FC;
  if (i >= total) return;
  int row = (int)(i >> 7);
  int c   = (int)(i & 127);
  h0[i] = (c < 96) ? x[(long)row * 96 + c]
                   : temb[(long)tid[row] * 32 + (c - 96)];
}

__global__ void k_deg_init(float* __restrict__ deg, int N) {
  int i = blockIdx.x * blockDim.x + threadIdx.x;
  if (i < N) deg[i] = 1.0f;              // self-loop contributes 1
}

__global__ void k_deg_edges(const int* __restrict__ dst, float* __restrict__ deg, int E) {
  int i = blockIdx.x * blockDim.x + threadIdx.x;
  if (i < E) atomicAdd(&deg[dst[i]], 1.0f);
}

__global__ void k_deg_recip(float* __restrict__ deg, int N) {
  int i = blockIdx.x * blockDim.x + threadIdx.x;
  if (i < N) deg[i] = 1.0f / deg[i];     // cnt >= 1 always
}

__global__ void k_zero(float* __restrict__ p, long n) {
  long i = (long)blockIdx.x * blockDim.x + threadIdx.x;
  if (i < n) p[i] = 0.0f;
}

// ---------------------------------------------------------------------------
// One wave32 per edge (8 edges / 256-thread block). Lane owns a float4 slice.
// e in [E, E+N) are the implicit self-loops with relation id 1.
// ---------------------------------------------------------------------------
__global__ void k_edge_msg(const int* __restrict__ src,
                           const int* __restrict__ dst,
                           const int* __restrict__ et,
                           const float* __restrict__ rel,   // [R, 128]
                           const float* __restrict__ h,     // [N, 128]
                           float* __restrict__ agg,         // [N, 128]
                           int E, int N) {
  int e    = blockIdx.x * 8 + (threadIdx.x >> 5);
  int lane = threadIdx.x & 31;
  if (e >= E + N) return;
  int s, d, t;
  if (e < E) { s = src[e]; d = dst[e]; t = et[e]; }
  else       { s = d = e - E; t = 1; }
  const float4 hv = ((const float4*)(h   + (long)s * FC))[lane];
  const float4 rv = ((const float4*)(rel + (long)t * FC))[lane];
  float* ap = agg + (long)d * FC + lane * 4;
  atomicAdd(ap + 0, hv.x * rv.x);
  atomicAdd(ap + 1, hv.y * rv.y);
  atomicAdd(ap + 2, hv.z * rv.z);
  atomicAdd(ap + 3, hv.w * rv.w);
}

// ---------------------------------------------------------------------------
// Pack [w_out; w_loop] (256 x 128 fp32) directly into the WMMA B-fragment
// layout as bf16. Linear index i = kt*4096 + j*512 + lane*16 + p, so each
// lane's 16 bf16 fragment elements are 32 contiguous bytes in LDS later.
//   base0 = (lane & 16) ? 8 : 0
//   k = kt*32 + (p < 8 ? base0 + p : 16 + base0 + (p - 8))   (ISA 7.12.2)
//   n = j*16 + (lane & 15)
// ---------------------------------------------------------------------------
__global__ void k_prep_w(const float* __restrict__ w_out,
                         const float* __restrict__ w_loop,
                         unsigned short* __restrict__ wcs) {
  int i = blockIdx.x * blockDim.x + threadIdx.x;   // 32768 threads total
  if (i >= 8 * 8 * 32 * 16) return;
  int p    = i & 15;
  int lane = (i >> 4) & 31;
  int j    = (i >> 9) & 7;
  int kt   = (i >> 12) & 7;
  int base0 = (lane & 16) ? 8 : 0;
  int k = kt * 32 + ((p < 8) ? (base0 + p) : (16 + base0 + (p - 8)));
  int n = j * 16 + (lane & 15);
  float v = (k < FC) ? w_out[(long)k * FC + n] : w_loop[(long)(k - FC) * FC + n];
  wcs[i] = __builtin_bit_cast(unsigned short, (__bf16)v);
}

// ---------------------------------------------------------------------------
// out = act( (agg * inv_deg) @ w_out + h @ w_loop )
// Fused as A[N,256] @ B[256,128] with A = [agg*inv_deg | h], B = [w_out; w_loop].
// Block = 256 threads = 8 waves; block tile = 128 rows x 128 cols.
// Pre-swizzled bf16 weights (64 KB) are staged to LDS once per block via the
// CDNA5 async global->LDS path (ASYNCcnt), then each B fragment is one
// 32-byte LDS read. v_wmma_f32_16x16x32_bf16, fp32 accumulate.
// ---------------------------------------------------------------------------
__global__ void k_gemm(const float* __restrict__ agg,
                       const float* __restrict__ invdeg,
                       const float* __restrict__ h,
                       const unsigned short* __restrict__ wcs, // swizzled bf16, 32768 elems
                       float* __restrict__ out,
                       int N, int relu) {
  __shared__ __align__(32) unsigned short lds_w[8 * 8 * 32 * 16]; // 64 KB

  // ---- stage swizzled weights into LDS (coalesced 16B per lane) ----
  {
    i4v* gw = (i4v*)wcs;                          // 4096 x 16B
    i4v* lw = (i4v*)lds_w;
#ifdef HAVE_ASYNC_LDS
#pragma unroll
    for (int t = 0; t < 16; ++t) {
      int idx = (int)threadIdx.x + t * 256;
      __builtin_amdgcn_global_load_async_to_lds_b128(
          (gas_i4p)(gw + idx), (las_i4p)(lw + idx), 0, 0);
    }
    wait_async_zero();
    __syncthreads();
#else
#pragma unroll
    for (int t = 0; t < 16; ++t) {
      int idx = (int)threadIdx.x + t * 256;
      lw[idx] = gw[idx];
    }
    __syncthreads();
#endif
  }

  const int lane = threadIdx.x & 31;
  const int wv   = threadIdx.x >> 5;
  const int r0   = blockIdx.x * 128 + wv * 16;

  // A-fragment addressing (ISA 7.12.2, 16-bit A 16x32):
  //   row M = lane & 15 ; lanes 0-15 hold K {0-7,16-23}, lanes 16-31 K {8-15,24-31}
  const int mA     = r0 + (lane & 15);
  const bool rowok = mA < N;
  const long rb    = (long)(rowok ? mA : 0) * FC;
  const float scAgg = rowok ? invdeg[rowok ? mA : 0] : 0.0f;
  const float scH   = rowok ? 1.0f : 0.0f;
  const int base0   = (lane & 16) ? 8 : 0;

  v8f acc[8] = {};

#pragma unroll 1
  for (int kt = 0; kt < 8; ++kt) {
    const int  kk    = kt * 32;              // global K base (0..224)
    const bool first = kt < 4;               // K < 128 -> agg part, else h part
    const float* srcp = first ? (agg + rb) : (h + rb);
    const int  ko    = first ? kk : (kk - 128);
    const float sc   = first ? scAgg : scH;

    const float4 p0 = *(const float4*)(srcp + ko + base0);
    const float4 p1 = *(const float4*)(srcp + ko + base0 + 4);
    const float4 p2 = *(const float4*)(srcp + ko + 16 + base0);
    const float4 p3 = *(const float4*)(srcp + ko + 16 + base0 + 4);

    float av[16];
    av[0]=p0.x*sc;  av[1]=p0.y*sc;  av[2]=p0.z*sc;  av[3]=p0.w*sc;
    av[4]=p1.x*sc;  av[5]=p1.y*sc;  av[6]=p1.z*sc;  av[7]=p1.w*sc;
    av[8]=p2.x*sc;  av[9]=p2.y*sc;  av[10]=p2.z*sc; av[11]=p2.w*sc;
    av[12]=p3.x*sc; av[13]=p3.y*sc; av[14]=p3.z*sc; av[15]=p3.w*sc;

    v16bf afrag;
#pragma unroll
    for (int p = 0; p < 16; ++p) afrag[p] = (__bf16)av[p];

#pragma unroll
    for (int j = 0; j < 8; ++j) {
      // one lane's fragment = 32 contiguous bytes in LDS
      const v16bf bfrag =
          *(const v16bf*)(lds_w + (((kt * 8 + j) * 32 + lane) << 4));
      acc[j] = __builtin_amdgcn_wmma_f32_16x16x32_bf16(
          false, afrag, false, bfrag, (short)0, acc[j], false, false);
    }
  }

  // C/D layout: VGPR r -> M = r (+8 for lanes 16-31), N = lane & 15.
  // Wave-uniform full/tail split: only the last block ever has partial rows,
  // so 390/391 blocks take the unguarded store path (no exec-mask juggling).
  const int mbase = r0 + ((lane & 16) ? 8 : 0);
  const int col0  = lane & 15;
  if (r0 + 16 <= N) {
#pragma unroll
    for (int j = 0; j < 8; ++j) {
      const int col = j * 16 + col0;
#pragma unroll
      for (int r = 0; r < 8; ++r) {
        float v = acc[j][r];
        if (relu) v = fmaxf(v, 0.0f);
        out[(long)(mbase + r) * FC + col] = v;
      }
    }
  } else {
#pragma unroll
    for (int j = 0; j < 8; ++j) {
      const int col = j * 16 + col0;
#pragma unroll
      for (int r = 0; r < 8; ++r) {
        const int mr = mbase + r;
        if (mr < N) {
          float v = acc[j][r];
          if (relu) v = fmaxf(v, 0.0f);
          out[(long)mr * FC + col] = v;
        }
      }
    }
  }
}

// ---------------------------------------------------------------------------
extern "C" void kernel_launch(void* const* d_in, const int* in_sizes, int n_in,
                              void* d_out, int out_size, void* d_ws, size_t ws_size,
                              hipStream_t stream) {
  const float* x     = (const float*)d_in[0];
  const int*   tid   = (const int*)  d_in[1];
  const int*   ei    = (const int*)  d_in[2];   // [2, E]: src then dst
  const int*   et    = (const int*)  d_in[3];
  const float* temb  = (const float*)d_in[4];
  const float* rel1  = (const float*)d_in[5];
  const float* wl1   = (const float*)d_in[6];
  const float* wo1   = (const float*)d_in[7];
  const float* rel2  = (const float*)d_in[8];
  const float* wl2   = (const float*)d_in[9];
  const float* wo2   = (const float*)d_in[10];
  const float* rel3  = (const float*)d_in[11];
  const float* wl3   = (const float*)d_in[12];
  const float* wo3   = (const float*)d_in[13];

  const int N = in_sizes[0] / 96;
  const int E = in_sizes[3];
  const int* src = ei;
  const int* dst = ei + E;

  auto alignup = [](size_t v) { return (v + 255) & ~(size_t)255; };
  char* w = (char*)d_ws;
  float* deg = (float*)w;  w += alignup(sizeof(float) * (size_t)N);
  float* hA  = (float*)w;  w += alignup(sizeof(float) * (size_t)N * FC);
  float* hB  = (float*)w;  w += alignup(sizeof(float) * (size_t)N * FC);
  float* agg = (float*)w;  w += alignup(sizeof(float) * (size_t)N * FC);
  unsigned short* wcs = (unsigned short*)w;   // swizzled bf16 weights, 64 KB

  const long NC = (long)N * FC;
  const int TB = 256;
  const int gNC  = (int)((NC + TB - 1) / TB);
  const int gN   = (N + TB - 1) / TB;
  const int gE   = (E + TB - 1) / TB;
  const int gEdg = (E + N + 7) / 8;          // 1 wave per edge, 8 waves/block
  const int gGem = (N + 127) / 128;
  const int gW   = (8 * 8 * 32 * 16) / TB;   // 128 blocks

  // h0 = concat(x, type_emb[tid])
  k_build_h0<<<gNC, TB, 0, stream>>>(x, tid, temb, hA, N);

  // inv_deg
  k_deg_init <<<gN, TB, 0, stream>>>(deg, N);
  k_deg_edges<<<gE, TB, 0, stream>>>(dst, deg, E);
  k_deg_recip<<<gN, TB, 0, stream>>>(deg, N);

  struct Layer { const float *rel, *wo, *wl, *hin; float* hout; int relu; };
  Layer layers[3] = {
    { rel1, wo1, wl1, hA, hB,            1 },
    { rel2, wo2, wl2, hB, hA,            1 },
    { rel3, wo3, wl3, hA, (float*)d_out, 0 },
  };

  for (int l = 0; l < 3; ++l) {
    const Layer& L = layers[l];
    k_zero    <<<gNC, TB, 0, stream>>>(agg, NC);
    k_edge_msg<<<gEdg, TB, 0, stream>>>(src, dst, et, L.rel, L.hin, agg, E, N);
    k_prep_w  <<<gW, TB, 0, stream>>>(L.wo, L.wl, wcs);
    k_gemm    <<<gGem, TB, 0, stream>>>(agg, deg, L.hin, wcs, L.hout, N, L.relu);
  }
}
